// DetectionModule_36215164239984
// MI455X (gfx1250) — matrix operand order repaired and checked by reading, not compiled, not run on hardware
//
#include <hip/hip_runtime.h>
#include <hip/hip_bf16.h>

typedef __bf16 v16bf __attribute__((ext_vector_type(16)));
typedef __bf16 v8bf  __attribute__((ext_vector_type(8)));
typedef float  v8f   __attribute__((ext_vector_type(8)));
typedef int    v4i_vs __attribute__((vector_size(4 * sizeof(int))));  // matches builtin param

#define WMMA_BF16(a, b, c) \
  __builtin_amdgcn_wmma_f32_16x16x32_bf16(false, (a), false, (b), (short)0, (c), false, false)

#define GLOBAL_AS __attribute__((address_space(1)))
#define LDS_AS    __attribute__((address_space(3)))

#if defined(__has_builtin)
#if __has_builtin(__builtin_amdgcn_global_load_async_to_lds_b128)
#define HAVE_ASYNC_LDS 1
#endif
#endif
#ifndef HAVE_ASYNC_LDS
#define HAVE_ASYNC_LDS 0
#endif

#if HAVE_ASYNC_LDS
#if __has_builtin(__builtin_amdgcn_s_wait_asynccnt)
#define WAIT_ASYNC() __builtin_amdgcn_s_wait_asynccnt(0)
#else
#define WAIT_ASYNC() asm volatile("s_wait_asynccnt 0x0" ::: "memory")
#endif
#else
#define WAIT_ASYNC() ((void)0)
#endif

// ---------------------------------------------------------------------------
// Prep: NCHW fp32 -> NHWC bf16 for feat [32,1280,16,16]
// ---------------------------------------------------------------------------
__global__ __launch_bounds__(256) void pack_feat_kernel(const float* __restrict__ in,
                                                        __bf16* __restrict__ out) {
  unsigned t  = blockIdx.x * 256u + threadIdx.x;   // 10,485,760 threads, exact
  unsigned p  = t & 255u;                          // spatial y*16+x
  unsigned bc = t >> 8;
  unsigned c  = bc % 1280u;
  unsigned b  = bc / 1280u;
  out[(size_t)(b * 256u + p) * 1280u + c] = (__bf16)in[t];
}

// ---------------------------------------------------------------------------
// Prep: fold BN scale into weights, repack OIHW fp32 -> [K][Cout] bf16 with
// k = tap*Cin + cin (tap-major so 32-wide K blocks never cross a tap).
// If hasBN==0, beta is the plain conv bias.
// ---------------------------------------------------------------------------
__global__ __launch_bounds__(256) void pack_w_kernel(
    const float* __restrict__ w, const float* __restrict__ g,
    const float* __restrict__ beta, const float* __restrict__ m,
    const float* __restrict__ v, __bf16* __restrict__ Wp,
    float* __restrict__ shift, int Cin, int Cout, int hasBN) {
  int t = blockIdx.x * 256 + threadIdx.x;
  int total = Cin * 9 * Cout;
  if (t >= total) return;
  int o   = t % Cout;
  int k   = t / Cout;
  int tap = k / Cin;
  int cin = k - tap * Cin;
  float scale = 1.f, sh;
  if (hasBN) {
    float s = g[o] * rsqrtf(v[o] + 1e-5f);
    scale = s;
    sh = beta[o] - m[o] * s;
  } else {
    sh = beta[o];
  }
  Wp[t] = (__bf16)(w[((size_t)o * Cin + cin) * 9 + tap] * scale);
  if (k == 0) shift[o] = sh;
}

// ---------------------------------------------------------------------------
// Implicit-GEMM 3x3 pad-1 conv on 16x16 feature maps, batch 32.
// GEMM: M=16 positions of one x-row, N=16 output channels, K=Cin*9.
//
// Block = 8 waves, ALL sharing one 32-cout group; each wave owns 2 position
// rows (2x2 WMMA register tile -> 4 v_wmma per K-step).
// Weights double-buffered through LDS in 256-K chunks via
// GLOBAL_LOAD_ASYNC_TO_LDS_B128 (ASYNCcnt) -> 8x cut in L2 weight traffic.
// Activation pointers are strength-reduced: +64B per K-step; base pointers
// and padding masks recomputed only at (rare, uniform) tap boundaries.
// Grid: (Cout/32) groups x 32 row-pair octets -> exact fit, EXEC all-ones.
// ---------------------------------------------------------------------------
__global__ __launch_bounds__(256) void conv3x3_wmma(
    const __bf16* __restrict__ X, const __bf16* __restrict__ Wp,
    const float* __restrict__ shift, __bf16* __restrict__ Y,
    int Cin, int Cout) {
  __shared__ __bf16 sWp[2][256 * 32];   // 2 x 16 KB: [k-in-chunk][32 couts]

  const int tid  = threadIdx.x;
  const int lane = tid & 31;
  const int wnum = tid >> 5;                 // wave in block: 0..7
  const int ng   = blockIdx.x >> 5;          // cout group (32 couts)
  const int mgrp = blockIdx.x & 31;
  const int coutbase = ng * 32;
  const int mg   = mgrp * 8 + wnum;          // 0..255 : pair of position rows
  const int pt0  = mg * 2;                   // global row tile (b*16 + y), even
  const int bb   = pt0 >> 4;
  const int y0   = pt0 & 15;
  const int xm   = lane & 15;                // A-fragment M = x position
  const int hi   = lane >> 4;
  const int off8 = hi * 8;                   // K sub-offset per ISA A layout

  const int K = Cin * 9;
  const int nChunk = (K + 255) >> 8;

  auto stage = [&](int c) {
    const int kc   = c << 8;
    const int rows = (K - kc < 256) ? (K - kc) : 256;
    if (tid < rows) {
      const __bf16* src = Wp + (size_t)(kc + tid) * Cout + coutbase;
      __bf16*       dst = &sWp[c & 1][tid * 32];
#if HAVE_ASYNC_LDS
      #pragma unroll
      for (int j = 0; j < 4; ++j)
        __builtin_amdgcn_global_load_async_to_lds_b128(
            (GLOBAL_AS v4i_vs*)(src + j * 8),
            (LDS_AS v4i_vs*)(dst + j * 8), 0, 0);
#else
      #pragma unroll
      for (int j = 0; j < 4; ++j)
        *(v8bf*)(dst + j * 8) = *(const v8bf*)(src + j * 8);
#endif
    }
  };

  v8f c00 = {}, c01 = {}, c10 = {}, c11 = {};
  int tap = 0, cinb = 0, dy = -1, dx = -1;
  int kb = 0;

  // Per-lane activation pointers + padding masks; refreshed only at tap change.
  const __bf16* pa0;
  const __bf16* pa1;
  bool ok0, ok1;
  auto setTap = [&]() {
    const int  xp  = xm + dx;
    const int  yp0 = y0 + dy;
    const bool xok = (unsigned)xp < 16u;
    ok0 = xok && ((unsigned)yp0 < 16u);
    ok1 = xok && ((unsigned)(yp0 + 1) < 16u);
    pa0 = X + ((ptrdiff_t)(bb * 256 + yp0 * 16 + xp) * Cin + off8);  // cinb == 0 here
    pa1 = pa0 + (ptrdiff_t)16 * Cin;                                  // next y row
  };
  setTap();

  stage(0);
  for (int c = 0; c < nChunk; ++c) {
    WAIT_ASYNC();
    __syncthreads();                 // chunk c resident; chunk c-1 fully read
    if (c + 1 < nChunk) stage(c + 1);

    const __bf16* buf = &sWp[c & 1][0];
    #pragma unroll
    for (int it = 0; it < 8; ++it) {
      if (kb >= K) break;

      v16bf a0 = {}, a1 = {};
      if (ok0) {
        const v8bf* p = (const v8bf*)pa0;
        v8bf lo = p[0], hh = p[2];  // +0 and +16 elements (ISA A K layout)
        #pragma unroll
        for (int i = 0; i < 8; ++i) { a0[i] = lo[i]; a0[i + 8] = hh[i]; }
      }
      if (ok1) {
        const v8bf* p = (const v8bf*)pa1;
        v8bf lo = p[0], hh = p[2];
        #pragma unroll
        for (int i = 0; i < 8; ++i) { a1[i] = lo[i]; a1[i + 8] = hh[i]; }
      }

      // B fragments from LDS: lane = N column (cout), elements = K rows (ISA table order)
      const __bf16* wrow = buf + (size_t)(it * 32 + off8) * 32 + (lane & 15);
      v16bf b0, b1;
      {
        #pragma unroll
        for (int i = 0; i < 8; ++i) { b0[i] = wrow[i * 32]; b0[i + 8] = wrow[(16 + i) * 32]; }
        #pragma unroll
        for (int i = 0; i < 8; ++i) { b1[i] = wrow[i * 32 + 16]; b1[i + 8] = wrow[(16 + i) * 32 + 16]; }
      }

      c00 = WMMA_BF16(a0, b0, c00);
      c01 = WMMA_BF16(a0, b1, c01);
      c10 = WMMA_BF16(a1, b0, c10);
      c11 = WMMA_BF16(a1, b1, c11);

      kb   += 32;
      cinb += 32;
      pa0  += 32;
      pa1  += 32;
      if (cinb == Cin) {
        cinb = 0;
        ++tap;
        int t3 = tap / 3;
        dy = t3 - 1;
        dx = (tap - 3 * t3) - 1;
        setTap();
      }
    }
  }

  // Epilogue: shift + SiLU + bf16 store. C/D layout: VGPR v -> M=v+8*hi, N=lane&15
  const int n = lane & 15;
  const float sh0 = shift[coutbase + n];
  const float sh1 = shift[coutbase + 16 + n];
  auto stt = [&](const v8f& cc, int row, int coutofs, float sh) {
    #pragma unroll
    for (int v = 0; v < 8; ++v) {
      float val = cc[v] + sh;
      float s = val * __builtin_amdgcn_rcpf(1.f + __expf(-val));   // SiLU, fast rcp
      Y[(size_t)(row * 16 + v + 8 * hi) * Cout + coutbase + coutofs + n] = (__bf16)s;
    }
  };
  stt(c00, pt0,     0,  sh0);
  stt(c01, pt0,     16, sh1);
  stt(c10, pt0 + 1, 0,  sh0);
  stt(c11, pt0 + 1, 16, sh1);
}

// ---------------------------------------------------------------------------
// Fused 1x1 convs (4x64 bbox, 1x32 obj) + sigmoid + box decode + mask.
// One thread per spatial position (8192 total).
// ---------------------------------------------------------------------------
__global__ __launch_bounds__(256) void decode_kernel(
    const __bf16* __restrict__ hb, const __bf16* __restrict__ ho,
    const float* __restrict__ wb2, const float* __restrict__ bb2,
    const float* __restrict__ wo2, const float* __restrict__ bo2,
    float* __restrict__ out) {
  int p = blockIdx.x * 256 + threadIdx.x;  // exact: 32 blocks * 256
  const __bf16* h = hb + (size_t)p * 64;
  float a0 = bb2[0], a1 = bb2[1], a2 = bb2[2], a3 = bb2[3];
  #pragma unroll 8
  for (int i = 0; i < 64; ++i) {
    float hv = (float)h[i];
    a0 += wb2[i]       * hv;
    a1 += wb2[64 + i]  * hv;
    a2 += wb2[128 + i] * hv;
    a3 += wb2[192 + i] * hv;
  }
  const __bf16* o = ho + (size_t)p * 32;
  float obj = bo2[0];
  #pragma unroll 8
  for (int i = 0; i < 32; ++i) obj += wo2[i] * (float)o[i];
  float conf = 1.f / (1.f + __expf(-obj));
  int x = p & 15, y = (p >> 4) & 15;
  float* bx = out + (size_t)p * 5;
  bx[0] = ((float)x + a0) * 16.f;   // IMG/W = 256/16
  bx[1] = ((float)y + a1) * 16.f;
  bx[2] = a2 * 16.f;
  bx[3] = a3 * 16.f;
  bx[4] = conf;
  out[40960 + p] = conf > 0.05f ? 1.f : 0.f;
}

// ---------------------------------------------------------------------------
extern "C" void kernel_launch(void* const* d_in, const int* in_sizes, int n_in,
                              void* d_out, int out_size, void* d_ws, size_t ws_size,
                              hipStream_t stream) {
  const float* feat = (const float*)d_in[0];
  const float* w1 = (const float*)d_in[1];  const float* g1 = (const float*)d_in[2];
  const float* b1 = (const float*)d_in[3];  const float* m1 = (const float*)d_in[4];
  const float* v1 = (const float*)d_in[5];
  const float* w2 = (const float*)d_in[6];  const float* g2 = (const float*)d_in[7];
  const float* b2 = (const float*)d_in[8];  const float* m2 = (const float*)d_in[9];
  const float* v2 = (const float*)d_in[10];
  const float* w3 = (const float*)d_in[11]; const float* g3 = (const float*)d_in[12];
  const float* b3 = (const float*)d_in[13]; const float* m3 = (const float*)d_in[14];
  const float* v3 = (const float*)d_in[15];
  const float* wb1 = (const float*)d_in[16]; const float* bb1 = (const float*)d_in[17];
  const float* wb2 = (const float*)d_in[18]; const float* bb2 = (const float*)d_in[19];
  const float* wo1 = (const float*)d_in[20]; const float* bo1 = (const float*)d_in[21];
  const float* wo2 = (const float*)d_in[22]; const float* bo2 = (const float*)d_in[23];

  char* ws = (char*)d_ws;
  size_t off = 0;
  auto alloc = [&](size_t bytes) -> char* {
    char* p = ws + off;
    off = (off + bytes + 255) & ~(size_t)255;
    return p;
  };
  __bf16* featN = (__bf16*)alloc(10485760ull * 2);  // [8192][1280]
  __bf16* Wp1   = (__bf16*)alloc(2211840ull * 2);
  float*  sh1   = (float*)alloc(192 * 4);
  __bf16* x1    = (__bf16*)alloc(1572864ull * 2);   // [8192][192]
  __bf16* Wp2   = (__bf16*)alloc(221184ull * 2);
  float*  sh2   = (float*)alloc(128 * 4);
  __bf16* x2    = (__bf16*)alloc(1048576ull * 2);   // [8192][128]
  __bf16* Wp3   = (__bf16*)alloc(110592ull * 2);
  float*  sh3   = (float*)alloc(96 * 4);
  __bf16* x3    = (__bf16*)alloc(786432ull * 2);    // [8192][96]
  __bf16* Wpb1  = (__bf16*)alloc(55296ull * 2);
  float*  shb1  = (float*)alloc(64 * 4);
  __bf16* hb    = (__bf16*)alloc(524288ull * 2);    // [8192][64]
  __bf16* Wpo1  = (__bf16*)alloc(27648ull * 2);
  float*  sho1  = (float*)alloc(32 * 4);
  __bf16* ho    = (__bf16*)alloc(262144ull * 2);    // [8192][32]
  (void)ws_size; (void)n_in; (void)in_sizes; (void)out_size;

  // --- prep ---
  pack_feat_kernel<<<40960, 256, 0, stream>>>(feat, featN);
  pack_w_kernel<<<8640, 256, 0, stream>>>(w1, g1, b1, m1, v1, Wp1, sh1, 1280, 192, 1);
  pack_w_kernel<<<864,  256, 0, stream>>>(w2, g2, b2, m2, v2, Wp2, sh2, 192, 128, 1);
  pack_w_kernel<<<432,  256, 0, stream>>>(w3, g3, b3, m3, v3, Wp3, sh3, 128, 96, 1);
  pack_w_kernel<<<216,  256, 0, stream>>>(wb1, nullptr, bb1, nullptr, nullptr, Wpb1, shb1, 96, 64, 0);
  pack_w_kernel<<<108,  256, 0, stream>>>(wo1, nullptr, bo1, nullptr, nullptr, Wpo1, sho1, 96, 32, 0);

  // --- conv stack (blocks = 32 * Cout/32, 8 waves each, exact fit) ---
  conv3x3_wmma<<<192, 256, 0, stream>>>(featN, Wp1, sh1, x1, 1280, 192);
  conv3x3_wmma<<<128, 256, 0, stream>>>(x1,   Wp2, sh2, x2, 192, 128);
  conv3x3_wmma<<<96,  256, 0, stream>>>(x2,   Wp3, sh3, x3, 128, 96);
  conv3x3_wmma<<<64,  256, 0, stream>>>(x3,   Wpb1, shb1, hb, 96, 64);
  conv3x3_wmma<<<32,  256, 0, stream>>>(x3,   Wpo1, sho1, ho, 96, 32);

  // --- heads + decode ---
  decode_kernel<<<32, 256, 0, stream>>>(hb, ho, wb2, bb2, wo2, bo2, (float*)d_out);
}
